// MultiHeadAttention_83949430768084
// MI455X (gfx1250) — matrix operand order repaired and checked by reading, not compiled, not run on hardware
//
#include <hip/hip_runtime.h>
#include <hip/hip_bf16.h>

// ---------------------------------------------------------------------------
// MHA on MI455X (gfx1250): bf16 WMMA (f32 accum) for all four GEMM stages,
// flash-attention with fully register-resident online softmax (lane-shuffle
// row reductions), so the S x S score tensor never exists anywhere.
// ---------------------------------------------------------------------------

#define EMBED 1024
#define HEADS 16
#define HD    64
#define BATCH 2
#define SEQ   2048
#define MTOT  (BATCH * SEQ)   // 4096 rows

typedef __attribute__((ext_vector_type(16))) __bf16 v16bf;
typedef __attribute__((ext_vector_type(8)))  float  v8f;

union FragU { v16bf v; unsigned int u[8]; };
union PairU { unsigned int u; __bf16 h[2]; };

__device__ __forceinline__ uint2 pack4_bf16(float4 f) {
  PairU a, b;
  a.h[0] = (__bf16)f.x; a.h[1] = (__bf16)f.y;   // -> v_cvt pair packing
  b.h[0] = (__bf16)f.z; b.h[1] = (__bf16)f.w;
  return make_uint2(a.u, b.u);
}

// A fragment: row-major [M][K] tile. rowptr = &tile[m][0] with m = lane%16.
// ISA 16-bit A layout: VGPR j in 0..3 holds K = kbase + 8*hi + 2j,2j+1;
//                      VGPR j in 4..7 holds K = kbase + 16 + 8*hi + ...
__device__ __forceinline__ v16bf a_frag_ld(const __bf16* rowptr, int kbase, int hi) {
  const unsigned int* rp = (const unsigned int*)rowptr;
  const int p0 = (kbase + 8 * hi) >> 1;
  FragU f;
#pragma unroll
  for (int j = 0; j < 4; ++j) f.u[j] = rp[p0 + j];
#pragma unroll
  for (int j = 0; j < 4; ++j) f.u[4 + j] = rp[p0 + 8 + j];
  return f.v;
}

// B fragment: row-major [K][N] tile (row stride 72). Lane column N = lane%16;
// elements 0..7 hold K = kbase + 8*hi + s, elements 8..15 K = kbase + 16 + 8*hi + (s-8).
// Caller passes p = &tile[kbase][nbase + lane%16].
__device__ __forceinline__ v16bf b_frag_ld(const __bf16* p, int hi) {
  v16bf f;
#pragma unroll
  for (int j = 0; j < 8; ++j) {
    f[j] = p[(8 * hi + j) * 72];
    f[8 + j] = p[(16 + 8 * hi + j) * 72];
  }
  return f;
}

__device__ __forceinline__ v8f wmma_bf16(v16bf a, v16bf b, v8f c) {
  return __builtin_amdgcn_wmma_f32_16x16x32_bf16(false, a, false, b,
                                                 (short)0, c, false, false);
}

// ---------------------------------------------------------------------------
// Kernel 1: per-head projection  out[b,h,s,d] = in[b,s,:] @ W[h,:,d] + bias
// Block = 128 threads (4 waves); 128-row x 64-col tile; 32 rows per wave so
// each staged B-fragment feeds two WMMAs (independent accumulator chains).
// ---------------------------------------------------------------------------
__global__ __launch_bounds__(128)
void qkv_proj_kernel(const float* __restrict__ in,    // [B*S, E]
                     const float* __restrict__ W,     // [H, E, HD]
                     const float* __restrict__ bias,  // [H, HD]
                     __bf16* __restrict__ out)        // [B, H, S, HD] bf16
{
  __shared__ __attribute__((aligned(16))) __bf16 Alds[128 * 36];  // [m][k]
  __shared__ __attribute__((aligned(16))) __bf16 Blds[32 * 72];   // [k][d]

  const int tid = threadIdx.x;
  const int lane = tid & 31, wave = tid >> 5;
  const int nl = lane & 15, hi = lane >> 4;
  const int m0 = blockIdx.x * 128, h = blockIdx.y;

  const int ar = tid >> 3, ac = (tid & 7) * 4;    // A slab: 8 float4 per row
  const int bk = tid >> 4, bc = (tid & 15) * 4;   // B slab: 16 float4 per row

  v8f acc[2][4] = {};

  for (int e0 = 0; e0 < EMBED; e0 += 32) {
#pragma unroll
    for (int v = 0; v < 8; ++v) {                 // 128x32 A slab, f32 -> bf16
      const int row = v * 16 + ar;
      float4 f = *(const float4*)(in + (size_t)(m0 + row) * EMBED + e0 + ac);
      *(uint2*)&Alds[row * 36 + ac] = pack4_bf16(f);
    }
#pragma unroll
    for (int v = 0; v < 4; ++v) {                 // 32x64 W slab
      const int kk = v * 8 + bk;
      float4 f = *(const float4*)(W + ((size_t)h * EMBED + e0 + kk) * HD + bc);
      *(uint2*)&Blds[kk * 72 + bc] = pack4_bf16(f);
    }
    if (e0 + 32 < EMBED)                          // gfx1250 global_prefetch_b8
      __builtin_prefetch(in + (size_t)(m0 + ar) * EMBED + e0 + 32, 0, 0);
    __syncthreads();

    const v16bf a0 = a_frag_ld(&Alds[(wave * 32 + nl) * 36], 0, hi);
    const v16bf a1 = a_frag_ld(&Alds[(wave * 32 + 16 + nl) * 36], 0, hi);
#pragma unroll
    for (int t = 0; t < 4; ++t) {
      const v16bf b = b_frag_ld(&Blds[t * 16 + nl], hi);
      acc[0][t] = wmma_bf16(a0, b, acc[0][t]);
      acc[1][t] = wmma_bf16(a1, b, acc[1][t]);
    }
    __syncthreads();
  }

#pragma unroll
  for (int s = 0; s < 2; ++s)
#pragma unroll
    for (int t = 0; t < 4; ++t) {
      const int d = t * 16 + nl;
      const float bv = bias[h * HD + d];
#pragma unroll
      for (int r = 0; r < 8; ++r) {
        const int g = m0 + wave * 32 + s * 16 + r + 8 * hi;  // = b*S + s
        const int b_ = g >> 11, sq = g & (SEQ - 1);
        out[(((size_t)b_ * HEADS + h) * SEQ + sq) * HD + d] =
            (__bf16)(acc[s][t][r] + bv);
      }
    }
}

// ---------------------------------------------------------------------------
// Kernel 2: flash attention per (b, h, 64-query tile). Streams 64-key tiles.
// Row statistics (max / sum / alpha) are computed entirely in registers with
// 16-lane __shfl_xor butterflies (the C/D WMMA layout keeps each row inside
// one 16-lane group). LDS is used only for K^T, V and the P lane-transpose.
// ---------------------------------------------------------------------------
__global__ __launch_bounds__(128)
void attn_kernel(const __bf16* __restrict__ Qb,   // [B,H,S,HD]
                 const __bf16* __restrict__ Kb,
                 const __bf16* __restrict__ Vb,
                 float* __restrict__ ctx)         // [B,S,E]
{
  __shared__ __attribute__((aligned(16))) __bf16 Kt[64 * 72];  // [d][key] transposed
  __shared__ __attribute__((aligned(16))) __bf16 Vl[64 * 72];  // [key][d]
  __shared__ __attribute__((aligned(16))) __bf16 Pl[64 * 72];  // [q][key] exp() bf16

  const int tid = threadIdx.x;
  const int lane = tid & 31, wave = tid >> 5;
  const int nl = lane & 15, hi = lane >> 4;
  const int q0 = blockIdx.x * 64, h = blockIdx.y, b = blockIdx.z;
  const size_t base = ((size_t)b * HEADS + h) * SEQ;

  // Q fragments live in registers for the whole block (16 rows x 64 d / wave)
  const __bf16* qrow = Qb + (base + q0 + wave * 16 + nl) * HD;
  const v16bf qf0 = a_frag_ld(qrow, 0, hi);
  const v16bf qf1 = a_frag_ld(qrow, 32, hi);

  v8f oacc[4] = {};
  float mreg[8], lreg[8];          // running row max / sum, rows r + 8*hi
#pragma unroll
  for (int r = 0; r < 8; ++r) { mreg[r] = -1e30f; lreg[r] = 0.0f; }

  for (int k0 = 0; k0 < SEQ; k0 += 64) {
    // --- stage V tile straight (b128 copies) ---
#pragma unroll
    for (int i = 0; i < 4; ++i) {
      const int idx = i * 128 + tid;               // 512 chunks of 8 bf16
      const int key = idx >> 3, c8 = (idx & 7) * 8;
      *(uint4*)&Vl[key * 72 + c8] =
          *(const uint4*)(Vb + (base + k0 + key) * HD + c8);
    }
    // --- stage K tile transposed: b64 reads, b16 scatter into Kt[d][key] ---
#pragma unroll
    for (int i = 0; i < 8; ++i) {
      const int idx = i * 128 + tid;               // 1024 chunks of 4 bf16
      const int key = idx >> 4, d = (idx & 15) * 4;
      const uint2 w = *(const uint2*)(Kb + (base + k0 + key) * HD + d);
      PairU p0, p1; p0.u = w.x; p1.u = w.y;
      Kt[(d + 0) * 72 + key] = p0.h[0];
      Kt[(d + 1) * 72 + key] = p0.h[1];
      Kt[(d + 2) * 72 + key] = p1.h[0];
      Kt[(d + 3) * 72 + key] = p1.h[1];
    }
    __syncthreads();

    // --- scores = (Q K^T) / sqrt(HD) ---
    v8f sacc[4] = {};
#pragma unroll
    for (int t = 0; t < 4; ++t) {
      sacc[t] = wmma_bf16(qf0, b_frag_ld(&Kt[t * 16 + nl], hi),        sacc[t]);
      sacc[t] = wmma_bf16(qf1, b_frag_ld(&Kt[32 * 72 + t * 16 + nl], hi), sacc[t]);
    }
#pragma unroll
    for (int t = 0; t < 4; ++t)
#pragma unroll
      for (int r = 0; r < 8; ++r) sacc[t][r] *= 0.125f;

    // --- online softmax, all in registers ---
    float alpha[8];
#pragma unroll
    for (int r = 0; r < 8; ++r) {
      float mx = fmaxf(fmaxf(sacc[0][r], sacc[1][r]),
                       fmaxf(sacc[2][r], sacc[3][r]));
#pragma unroll
      for (int m = 1; m <= 8; m <<= 1)             // stays inside 16-lane group
        mx = fmaxf(mx, __shfl_xor(mx, m, 32));
      const float mn = fmaxf(mreg[r], mx);
      alpha[r] = __expf(mreg[r] - mn);
      mreg[r] = mn;
    }
    const int prow0 = wave * 16 + 8 * hi;
#pragma unroll
    for (int r = 0; r < 8; ++r) {
      const float p0 = __expf(sacc[0][r] - mreg[r]);
      const float p1 = __expf(sacc[1][r] - mreg[r]);
      const float p2 = __expf(sacc[2][r] - mreg[r]);
      const float p3 = __expf(sacc[3][r] - mreg[r]);
      __bf16* pr = &Pl[(prow0 + r) * 72 + nl];     // own-wave rows only
      pr[0]  = (__bf16)p0;
      pr[16] = (__bf16)p1;
      pr[32] = (__bf16)p2;
      pr[48] = (__bf16)p3;
      float s = (p0 + p1) + (p2 + p3);
#pragma unroll
      for (int m = 1; m <= 8; m <<= 1)
        s += __shfl_xor(s, m, 32);
      lreg[r] = lreg[r] * alpha[r] + s;
    }
    // --- rescale O, then O += P @ V (P rows are this wave's own: no barrier,
    //     same-wave DS ops are in-order) ---
#pragma unroll
    for (int t = 0; t < 4; ++t)
#pragma unroll
      for (int r = 0; r < 8; ++r) oacc[t][r] *= alpha[r];

    const __bf16* prow = &Pl[(wave * 16 + nl) * 72];
    const v16bf pa0 = a_frag_ld(prow, 0, hi);
    const v16bf pa1 = a_frag_ld(prow, 32, hi);
#pragma unroll
    for (int t = 0; t < 4; ++t) {
      oacc[t] = wmma_bf16(pa0, b_frag_ld(&Vl[t * 16 + nl], hi),        oacc[t]);
      oacc[t] = wmma_bf16(pa1, b_frag_ld(&Vl[32 * 72 + t * 16 + nl], hi), oacc[t]);
    }
    __syncthreads();
  }

  // --- finalize: O /= l, store f32 into the concatenated-head ctx buffer ---
  float inv[8];
#pragma unroll
  for (int r = 0; r < 8; ++r) inv[r] = 1.0f / lreg[r];
#pragma unroll
  for (int t = 0; t < 4; ++t) {
    const int d = t * 16 + nl;
#pragma unroll
    for (int r = 0; r < 8; ++r) {
      const int row = wave * 16 + r + 8 * hi;
      ctx[((size_t)b * SEQ + q0 + row) * EMBED + h * HD + d] = oacc[t][r] * inv[r];
    }
  }
}

// ---------------------------------------------------------------------------
// Kernel 3: output projection  out = ctx @ Wo + bo  (4096x1024 x 1024x1024)
// 128-row x 64-col tile per block, 32 rows per wave.
// ---------------------------------------------------------------------------
__global__ __launch_bounds__(128)
void out_proj_kernel(const float* __restrict__ ctx, // [M, E]
                     const float* __restrict__ Wo,  // [E, E]
                     const float* __restrict__ bo,  // [E]
                     float* __restrict__ out)       // [M, E]
{
  __shared__ __attribute__((aligned(16))) __bf16 Alds[128 * 36];
  __shared__ __attribute__((aligned(16))) __bf16 Blds[32 * 72];

  const int tid = threadIdx.x;
  const int lane = tid & 31, wave = tid >> 5;
  const int nl = lane & 15, hi = lane >> 4;
  const int m0 = blockIdx.x * 128, n0 = blockIdx.y * 64;

  const int ar = tid >> 3, ac = (tid & 7) * 4;
  const int bk = tid >> 4, bc = (tid & 15) * 4;

  v8f acc[2][4] = {};
  for (int e0 = 0; e0 < EMBED; e0 += 32) {
#pragma unroll
    for (int v = 0; v < 8; ++v) {
      const int row = v * 16 + ar;
      float4 f = *(const float4*)(ctx + (size_t)(m0 + row) * EMBED + e0 + ac);
      *(uint2*)&Alds[row * 36 + ac] = pack4_bf16(f);
    }
#pragma unroll
    for (int v = 0; v < 4; ++v) {
      const int kk = v * 8 + bk;
      float4 f = *(const float4*)(Wo + (size_t)(e0 + kk) * EMBED + n0 + bc);
      *(uint2*)&Blds[kk * 72 + bc] = pack4_bf16(f);
    }
    if (e0 + 32 < EMBED)
      __builtin_prefetch(ctx + (size_t)(m0 + ar) * EMBED + e0 + 32, 0, 0);
    __syncthreads();

    const v16bf a0 = a_frag_ld(&Alds[(wave * 32 + nl) * 36], 0, hi);
    const v16bf a1 = a_frag_ld(&Alds[(wave * 32 + 16 + nl) * 36], 0, hi);
#pragma unroll
    for (int t = 0; t < 4; ++t) {
      const v16bf bb = b_frag_ld(&Blds[t * 16 + nl], hi);
      acc[0][t] = wmma_bf16(a0, bb, acc[0][t]);
      acc[1][t] = wmma_bf16(a1, bb, acc[1][t]);
    }
    __syncthreads();
  }

#pragma unroll
  for (int s = 0; s < 2; ++s)
#pragma unroll
    for (int t = 0; t < 4; ++t) {
      const int d = n0 + t * 16 + nl;
      const float bv = bo[d];
#pragma unroll
      for (int r = 0; r < 8; ++r) {
        const int row = m0 + wave * 32 + s * 16 + r + 8 * hi;
        out[(size_t)row * EMBED + d] = acc[s][t][r] + bv;
      }
    }
}

// ---------------------------------------------------------------------------
extern "C" void kernel_launch(void* const* d_in, const int* in_sizes, int n_in,
                              void* d_out, int out_size, void* d_ws, size_t ws_size,
                              hipStream_t stream) {
  const float* q  = (const float*)d_in[0];
  const float* k  = (const float*)d_in[1];
  const float* v  = (const float*)d_in[2];
  const float* Wq = (const float*)d_in[3];
  const float* Wk = (const float*)d_in[4];
  const float* Wv = (const float*)d_in[5];
  const float* bq = (const float*)d_in[6];
  const float* bk = (const float*)d_in[7];
  const float* bv = (const float*)d_in[8];
  const float* Wo = (const float*)d_in[9];
  const float* bo = (const float*)d_in[10];
  float* out = (float*)d_out;

  // Workspace: Q,K,V bf16 [B,H,S,HD] (8 MB each), f32 ctx [B,S,E] (16 MB).
  const size_t qkv_elems = (size_t)BATCH * HEADS * SEQ * HD;
  __bf16* Qb = (__bf16*)d_ws;
  __bf16* Kb = Qb + qkv_elems;
  __bf16* Vb = Kb + qkv_elems;
  float* ctx = (float*)(Vb + qkv_elems);

  const dim3 blk(128);  // 4 wave32 waves
  qkv_proj_kernel<<<dim3(MTOT / 128, HEADS), blk, 0, stream>>>(q, Wq, bq, Qb);
  qkv_proj_kernel<<<dim3(MTOT / 128, HEADS), blk, 0, stream>>>(k, Wk, bk, Kb);
  qkv_proj_kernel<<<dim3(MTOT / 128, HEADS), blk, 0, stream>>>(v, Wv, bv, Vb);
  attn_kernel<<<dim3(SEQ / 64, HEADS, BATCH), blk, 0, stream>>>(Qb, Kb, Vb, ctx);
  out_proj_kernel<<<dim3(MTOT / 128, EMBED / 64), blk, 0, stream>>>(ctx, Wo, bo, out);
}
